// IdeficsAttention_13812614824415
// MI455X (gfx1250) — compile-verified, matrix operand order checked by reading, not run on hardware
//
#include <hip/hip_runtime.h>

typedef unsigned short u16;
typedef __bf16 bf16;
typedef __attribute__((ext_vector_type(16))) bf16  v16bf;
typedef __attribute__((ext_vector_type(8)))  float v8f;

#define HIDDEN 2048
#define NHEADS 16
#define HDIM   128
#define SEQ    2048

__device__ __forceinline__ u16 f2bf(float f) {
  unsigned u = __float_as_uint(f);
  u += 0x7fffu + ((u >> 16) & 1u);          // round-to-nearest-even
  return (u16)(u >> 16);
}
__device__ __forceinline__ float bf2f(u16 b) {
  return __uint_as_float(((unsigned)b) << 16);
}

struct Frag { union { v16bf v; uint4 q[2]; }; };

// ---------------------------------------------------------------- convert
__global__ void cvt_bf16(const float* __restrict__ in, u16* __restrict__ out, int n) {
  int i = blockIdx.x * blockDim.x + threadIdx.x;
  if (i < n) out[i] = f2bf(in[i]);
}

// ---------------------------------------------------------------- GEMM  C = A(MxK) * W(NxK)^T
// block tile 128x128, K-step 32, 8 waves each 32(M)x64(N) = 2x4 wmma tiles.
// Register double-buffering: next K-step's global loads issue before this
// step's WMMAs, so s_wait_loadcnt lands behind 8 WMMAs of compute.
template<bool BF16OUT>
__launch_bounds__(256)
__global__ void gemm_awt(const u16* __restrict__ A, const u16* __restrict__ W,
                         void* __restrict__ Cv, int N, int K) {
  __shared__ u16 As[128 * 40];   // 32 cols + pad 8 (keeps 16B align, kills bank conflicts)
  __shared__ u16 Ws[128 * 40];

  const int tid  = threadIdx.x;
  const int lane = tid & 31;
  const int wv   = tid >> 5;
  const int hi   = lane >> 4, ln = lane & 15;
  const int wm   = (wv & 3) * 32, wn = (wv >> 2) * 64;
  const int m0   = blockIdx.y * 128, n0 = blockIdx.x * 128;

  v8f acc[2][4];
  #pragma unroll
  for (int mi = 0; mi < 2; ++mi)
    #pragma unroll
    for (int ni = 0; ni < 4; ++ni)
      #pragma unroll
      for (int r = 0; r < 8; ++r) acc[mi][ni][r] = 0.f;

  const int ar0 = tid >> 2;              // 0..63
  const int ac0 = (tid & 3) * 8;         // 0,8,16,24  (elements)

  const u16* ap0 = A + (size_t)(m0 + ar0)      * K + ac0;
  const u16* ap1 = A + (size_t)(m0 + ar0 + 64) * K + ac0;
  const u16* wp0 = W + (size_t)(n0 + ar0)      * K + ac0;
  const u16* wp1 = W + (size_t)(n0 + ar0 + 64) * K + ac0;

  uint4 ra0 = *(const uint4*)ap0;
  uint4 ra1 = *(const uint4*)ap1;
  uint4 rw0 = *(const uint4*)wp0;
  uint4 rw1 = *(const uint4*)wp1;

  for (int kb = 0; kb < K; kb += 32) {
    __syncthreads();
    *(uint4*)(As + ar0 * 40 + ac0)        = ra0;
    *(uint4*)(As + (ar0 + 64) * 40 + ac0) = ra1;
    *(uint4*)(Ws + ar0 * 40 + ac0)        = rw0;
    *(uint4*)(Ws + (ar0 + 64) * 40 + ac0) = rw1;
    __syncthreads();

    if (kb + 32 < K) {                   // prefetch next K-step (latency hidden by WMMAs below)
      ra0 = *(const uint4*)(ap0 + kb + 32);
      ra1 = *(const uint4*)(ap1 + kb + 32);
      rw0 = *(const uint4*)(wp0 + kb + 32);
      rw1 = *(const uint4*)(wp1 + kb + 32);
    }

    const int c0 = hi * 8;               // lanes<16: K {0..7,16..23}; lanes>=16: {8..15,24..31}
    Frag af[2], wf[4];
    #pragma unroll
    for (int mi = 0; mi < 2; ++mi) {
      const u16* p = As + (wm + mi * 16 + ln) * 40 + c0;
      af[mi].q[0] = *(const uint4*)p;
      af[mi].q[1] = *(const uint4*)(p + 16);
    }
    #pragma unroll
    for (int ni = 0; ni < 4; ++ni) {
      const u16* p = Ws + (wn + ni * 16 + ln) * 40 + c0;
      wf[ni].q[0] = *(const uint4*)p;
      wf[ni].q[1] = *(const uint4*)(p + 16);
    }
    #pragma unroll
    for (int mi = 0; mi < 2; ++mi)
      #pragma unroll
      for (int ni = 0; ni < 4; ++ni)
        acc[mi][ni] = __builtin_amdgcn_wmma_f32_16x16x32_bf16(
            false, af[mi].v, false, wf[ni].v, (short)0, acc[mi][ni], false, false);
  }

  #pragma unroll
  for (int mi = 0; mi < 2; ++mi)
    #pragma unroll
    for (int ni = 0; ni < 4; ++ni)
      #pragma unroll
      for (int r = 0; r < 8; ++r) {
        size_t row = (size_t)(m0 + wm + mi * 16 + hi * 8 + r);
        size_t col = (size_t)(n0 + wn + ni * 16 + ln);
        float x = acc[mi][ni][r];
        if constexpr (BF16OUT) ((u16*)Cv)[row * N + col] = f2bf(x);
        else                   ((float*)Cv)[row * N + col] = x;
      }
}

// ---------------------------------------------------------------- RoPE + RMSNorm (+V transpose)
__launch_bounds__(128)
__global__ void rope_norm(const u16* __restrict__ qkv, const int* __restrict__ pos,
                          const float* __restrict__ qw, const float* __restrict__ kw,
                          u16* __restrict__ qo, u16* __restrict__ ko, u16* __restrict__ vt) {
  __shared__ float sq[128], sk[128], red[128];
  const int d = threadIdx.x;
  const int t = blockIdx.x >> 4;          // token 0..4095
  const int h = blockIdx.x & 15;
  const int b = t >> 11, s = t & 2047;

  size_t base = (size_t)t * (3 * HIDDEN) + h * HDIM + d;
  float qv = bf2f(qkv[base]);
  float kv = bf2f(qkv[base + HIDDEN]);
  float vv = bf2f(qkv[base + 2 * HIDDEN]);
  sq[d] = qv; sk[d] = kv;
  __syncthreads();

  int   p   = pos[t];
  int   i   = d & 63;
  float ang = (float)p * __powf(10000.f, -(float)i * (1.f / 64.f));
  float sn, cs; __sincosf(ang, &sn, &cs);
  float qr, kr;
  if (d < 64) { qr = sq[d] * cs - sq[d + 64] * sn; kr = sk[d] * cs - sk[d + 64] * sn; }
  else        { qr = sq[d] * cs + sq[d - 64] * sn; kr = sk[d] * cs + sk[d - 64] * sn; }

  red[d] = qr * qr; __syncthreads();
  for (int o = 64; o > 0; o >>= 1) { if (d < o) red[d] += red[d + o]; __syncthreads(); }
  float qn = qr * rsqrtf(red[0] * (1.f / 128.f) + 1e-6f) * qw[d] * 0.08838834764831845f;
  __syncthreads();
  red[d] = kr * kr; __syncthreads();
  for (int o = 64; o > 0; o >>= 1) { if (d < o) red[d] += red[d + o]; __syncthreads(); }
  float kn = kr * rsqrtf(red[0] * (1.f / 128.f) + 1e-6f) * kw[d];

  size_t ob = ((size_t)(b * NHEADS + h) * SEQ + s) * HDIM + d;
  qo[ob] = f2bf(qn);
  ko[ob] = f2bf(kn);
  vt[((size_t)(b * NHEADS + h) * HDIM + d) * SEQ + s] = f2bf(vv);   // V transposed [B,H,D,S]
}

// ---------------------------------------------------------------- flash attention (1 wave = 16 queries)
__launch_bounds__(256)
__global__ void flash_attn(const u16* __restrict__ Q, const u16* __restrict__ Km,
                           const u16* __restrict__ Vt, u16* __restrict__ Out) {
  __shared__ u16 Pt[8 * 16 * 40];
  const int lane = threadIdx.x & 31;
  const int wv   = threadIdx.x >> 5;
  const int unit = blockIdx.x * 8 + wv;   // 4096 units = (b,h,qtile)
  const int qt   = unit & 127;
  const int bh   = unit >> 7;
  const int b    = bh >> 4, h = bh & 15;
  const int q0   = qt * 16;
  const int hi   = lane >> 4, ln = lane & 15;

  const u16* qp = Q  + (size_t)bh * (SEQ * HDIM);
  const u16* kp = Km + (size_t)bh * (SEQ * HDIM);
  const u16* vp = Vt + (size_t)bh * (HDIM * SEQ);
  u16* pw = Pt + wv * (16 * 40);

  Frag qf[4];
  #pragma unroll
  for (int kk = 0; kk < 4; ++kk) {
    const u16* p = qp + (size_t)(q0 + ln) * HDIM + kk * 32 + hi * 8;
    qf[kk].q[0] = *(const uint4*)p;
    qf[kk].q[1] = *(const uint4*)(p + 16);
  }

  v8f o[8]; float mr[8], lr[8];
  #pragma unroll
  for (int j = 0; j < 8; ++j)
    #pragma unroll
    for (int r = 0; r < 8; ++r) o[j][r] = 0.f;
  #pragma unroll
  for (int r = 0; r < 8; ++r) { mr[r] = -3.0e38f; lr[r] = 0.f; }

  for (int kb = 0; kb < SEQ; kb += 32) {
    v8f s0, s1;
    #pragma unroll
    for (int r = 0; r < 8; ++r) { s0[r] = 0.f; s1[r] = 0.f; }

    #pragma unroll
    for (int kk = 0; kk < 4; ++kk) {                      // S = Q * K^T
      Frag k0, k1;
      const u16* p0 = kp + (size_t)(kb + ln) * HDIM + kk * 32 + hi * 8;
      k0.q[0] = *(const uint4*)p0; k0.q[1] = *(const uint4*)(p0 + 16);
      const u16* p1 = kp + (size_t)(kb + 16 + ln) * HDIM + kk * 32 + hi * 8;
      k1.q[0] = *(const uint4*)p1; k1.q[1] = *(const uint4*)(p1 + 16);
      s0 = __builtin_amdgcn_wmma_f32_16x16x32_bf16(false, qf[kk].v, false, k0.v, (short)0, s0, false, false);
      s1 = __builtin_amdgcn_wmma_f32_16x16x32_bf16(false, qf[kk].v, false, k1.v, (short)0, s1, false, false);
    }

    #pragma unroll
    for (int r = 0; r < 8; ++r) {                         // online softmax, rows r+8*hi
      float mv = fmaxf(s0[r], s1[r]);
      #pragma unroll
      for (int off = 1; off < 16; off <<= 1) mv = fmaxf(mv, __shfl_xor(mv, off, 32));
      float mn    = fmaxf(mr[r], mv);
      float alpha = __expf(mr[r] - mn);
      mr[r] = mn;
      float p0v = __expf(s0[r] - mn);
      float p1v = __expf(s1[r] - mn);
      float rs  = p0v + p1v;
      #pragma unroll
      for (int off = 1; off < 16; off <<= 1) rs += __shfl_xor(rs, off, 32);
      lr[r] = lr[r] * alpha + rs;
      s0[r] = p0v; s1[r] = p1v;
      #pragma unroll
      for (int j = 0; j < 8; ++j) o[j][r] *= alpha;
    }

    #pragma unroll
    for (int r = 0; r < 8; ++r) {                         // P -> LDS (D-layout -> A-layout)
      int row = hi * 8 + r;
      pw[row * 40 + ln]      = f2bf(s0[r]);
      pw[row * 40 + 16 + ln] = f2bf(s1[r]);
    }
    Frag pf;
    { const u16* p = pw + ln * 40 + hi * 8;
      pf.q[0] = *(const uint4*)p; pf.q[1] = *(const uint4*)(p + 16); }

    #pragma unroll
    for (int j = 0; j < 8; ++j) {                         // O += P * V
      Frag vf;
      const u16* p = vp + (size_t)(j * 16 + ln) * SEQ + kb + hi * 8;
      vf.q[0] = *(const uint4*)p; vf.q[1] = *(const uint4*)(p + 16);
      o[j] = __builtin_amdgcn_wmma_f32_16x16x32_bf16(false, pf.v, false, vf.v, (short)0, o[j], false, false);
    }
  }

  #pragma unroll
  for (int r = 0; r < 8; ++r) {
    float inv  = 1.f / lr[r];
    int   srow = q0 + hi * 8 + r;
    size_t rbase = ((size_t)(b * SEQ + srow)) * HIDDEN + h * HDIM + ln;
    #pragma unroll
    for (int j = 0; j < 8; ++j)
      Out[rbase + j * 16] = f2bf(o[j][r] * inv);
  }
}

// ---------------------------------------------------------------- launch
extern "C" void kernel_launch(void* const* d_in, const int* in_sizes, int n_in,
                              void* d_out, int out_size, void* d_ws, size_t ws_size,
                              hipStream_t stream) {
  (void)in_sizes; (void)n_in; (void)out_size; (void)ws_size;
  const float* hs   = (const float*)d_in[0];
  const float* wqkv = (const float*)d_in[1];
  const float* wo   = (const float*)d_in[2];
  const float* qlw  = (const float*)d_in[3];
  const float* klw  = (const float*)d_in[4];
  const int*   pos  = (const int*)d_in[5];

  char* ws = (char*)d_ws;
  size_t off = 0;
  auto take = [&](size_t elems) {
    u16* p = (u16*)(ws + off);
    off += ((elems * 2 + 255) & ~(size_t)255);
    return p;
  };
  const size_t M = 4096;                       // B*S tokens
  u16* hs_bf   = take(M * 2048);
  u16* wqkv_bf = take((size_t)6144 * 2048);
  u16* wo_bf   = take((size_t)2048 * 2048);
  u16* qkv_bf  = take(M * 6144);
  u16* q_bf    = take(M * 2048);
  u16* k_bf    = take(M * 2048);
  u16* vt_bf   = take(M * 2048);
  u16* attn_bf = take(M * 2048);

  cvt_bf16<<<(int)((M * 2048) / 256), 256, 0, stream>>>(hs,   hs_bf,   (int)(M * 2048));
  cvt_bf16<<<(6144 * 2048) / 256,      256, 0, stream>>>(wqkv, wqkv_bf, 6144 * 2048);
  cvt_bf16<<<(2048 * 2048) / 256,      256, 0, stream>>>(wo,   wo_bf,   2048 * 2048);

  gemm_awt<true><<<dim3(6144 / 128, 4096 / 128), 256, 0, stream>>>(hs_bf, wqkv_bf, qkv_bf, 6144, 2048);
  rope_norm<<<4096 * 16, 128, 0, stream>>>(qkv_bf, pos, qlw, klw, q_bf, k_bf, vt_bf);
  flash_attn<<<512, 256, 0, stream>>>(q_bf, k_bf, vt_bf, attn_bf);
  gemm_awt<false><<<dim3(2048 / 128, 4096 / 128), 256, 0, stream>>>(attn_bf, wo_bf, d_out, 2048, 2048);
}
